// AttentionPooling_73744588473097
// MI455X (gfx1250) — matrix operand order repaired and checked by reading, not compiled, not run on hardware
//
#include <hip/hip_runtime.h>

typedef __bf16 bf16;
typedef __attribute__((ext_vector_type(16))) __bf16 v16bf;
typedef __attribute__((ext_vector_type(8)))  __bf16 v8bf;
typedef __attribute__((ext_vector_type(8)))  float  v8f;

#define HIDN  1024
#define HEADS 8
#define HDIM  128
#define BATCH 4
#define SEQ   2048
#define MROWS (BATCH*SEQ)   // 8192

// ---------------- helpers ----------------

__device__ __forceinline__ bf16 f2bf(float f) {
    union { float f; unsigned u; } x; x.f = f;
    unsigned r = x.u + 0x7FFFu + ((x.u >> 16) & 1u);   // round-nearest-even
    unsigned short s = (unsigned short)(r >> 16);
    bf16 o; __builtin_memcpy(&o, &s, 2); return o;
}

__device__ __forceinline__ v16bf cat8(v8bf lo, v8bf hi) {
    return __builtin_shufflevector(lo, hi, 0,1,2,3,4,5,6,7,8,9,10,11,12,13,14,15);
}

__device__ __forceinline__ v8f zero8() {
    v8f z = {0.f,0.f,0.f,0.f,0.f,0.f,0.f,0.f}; return z;
}

// A fragment (16xK=32, bf16). Per ISA 16-bit A layout:
// lanes 0-15 : row = lane,    K = k0+[0..7]  and k0+[16..23]
// lanes 16-31: row = lane-16, K = k0+[8..15] and k0+[24..31]
__device__ __forceinline__ v16bf load_a_frag(const bf16* base, int ld, int k0) {
    int l  = threadIdx.x & 31;
    int hi = l >> 4;
    const bf16* p = base + (size_t)(l & 15) * ld + k0 + hi * 8;
    v8bf lo = *(const v8bf*)p;
    v8bf hh = *(const v8bf*)(p + 16);
    return cat8(lo, hh);
}

// B fragment (K=32 x 16, bf16). Column n of B = row n of `base` (K-contiguous):
// lanes 0-15 : col = lane,    K = k0+[0..7]  and k0+[16..23]
// lanes 16-31: col = lane-16, K = k0+[8..15] and k0+[24..31]
__device__ __forceinline__ v16bf load_b_frag(const bf16* base, int ld, int k0) {
    int l  = threadIdx.x & 31;
    int hi = l >> 4;
    const bf16* p = base + (size_t)(l & 15) * ld + k0 + hi * 8;
    v8bf lo = *(const v8bf*)p;
    v8bf hh = *(const v8bf*)(p + 16);
    return cat8(lo, hh);
}

__device__ __forceinline__ v8f wmma_bf16(v16bf a, v16bf b, v8f c) {
    return __builtin_amdgcn_wmma_f32_16x16x32_bf16(false, a, false, b, (short)0, c, false, false);
}

// Async 16-byte copy global -> LDS (CDNA5 GLOBAL_LOAD_ASYNC_TO_LDS_B128,
// tracked by ASYNCcnt). LDS byte offset = low 32 bits of the generic pointer.
__device__ __forceinline__ void async_load16(const bf16* g, bf16* l) {
    unsigned           lds = (unsigned)(uintptr_t)l;
    unsigned long long ga  = (unsigned long long)(uintptr_t)g;
    asm volatile("global_load_async_to_lds_b128 %0, %1, off"
                 :: "v"(lds), "v"(ga) : "memory");
}

template<int N> __device__ __forceinline__ void wait_async() {
    asm volatile("s_wait_asynccnt %0" :: "n"(N) : "memory");
}

// ---------------- kernel 1: fp32 -> bf16 conversion + zero pooled acc ----------------

__global__ __launch_bounds__(256) void convert_kernel(
    const float* __restrict__ X,
    const float* __restrict__ Wq, const float* __restrict__ Wk,
    const float* __restrict__ Wv, const float* __restrict__ Wo,
    bf16* __restrict__ Xb, bf16* __restrict__ Wqb, bf16* __restrict__ Wkb,
    bf16* __restrict__ Wvb, bf16* __restrict__ Wob, float* __restrict__ accPool)
{
    int i = blockIdx.x * blockDim.x + threadIdx.x;
    int stride = gridDim.x * blockDim.x;
    for (int t = i; t < MROWS * HIDN; t += stride) Xb[t] = f2bf(X[t]);
    for (int t = i; t < HIDN * HIDN; t += stride) {
        Wqb[t] = f2bf(Wq[t]); Wkb[t] = f2bf(Wk[t]);
        Wvb[t] = f2bf(Wv[t]); Wob[t] = f2bf(Wo[t]);
    }
    for (int t = i; t < BATCH * HIDN; t += stride) accPool[t] = 0.0f;
}

// ---------------- kernel 2: QKV projection GEMM (y = X * W^T + b) ----------------
// Q,K stored [B,H,N,D]; V stored TRANSPOSED [B,H,D,N] so that the attention
// P*V B-fragments become contiguous-K loads. Q gets 1/sqrt(HDIM) folded in.

__global__ __launch_bounds__(256) void qkv_kernel(
    const bf16* __restrict__ Xb,
    const bf16* __restrict__ Wqb, const bf16* __restrict__ Wkb, const bf16* __restrict__ Wvb,
    const float* __restrict__ bq, const float* __restrict__ bk, const float* __restrict__ bv,
    bf16* __restrict__ Q, bf16* __restrict__ K, bf16* __restrict__ Vt)
{
    int sel = blockIdx.z;
    const bf16*  W    = (sel == 0) ? Wqb : (sel == 1) ? Wkb : Wvb;
    const float* bias = (sel == 0) ? bq  : (sel == 1) ? bk  : bv;
    bf16*        out  = (sel == 0) ? Q   : (sel == 1) ? K   : Vt;
    float scale = (sel == 0) ? 0.08838834764831845f : 1.0f;   // 1/sqrt(128)

    int w    = threadIdx.x >> 5;
    int tile = blockIdx.x * 8 + w;
    int gr0  = (tile >> 4) * 32;   // 16 N-tiles
    int oc0  = (tile & 15) * 64;

    v8f acc[2][4];
    #pragma unroll
    for (int r = 0; r < 2; ++r)
        #pragma unroll
        for (int c = 0; c < 4; ++c) acc[r][c] = zero8();

    for (int k0 = 0; k0 < HIDN; k0 += 32) {
        v16bf a0 = load_a_frag(Xb + (size_t)gr0 * HIDN,        HIDN, k0);
        v16bf a1 = load_a_frag(Xb + (size_t)(gr0 + 16) * HIDN, HIDN, k0);
        #pragma unroll
        for (int ct = 0; ct < 4; ++ct) {
            v16bf b = load_b_frag(W + (size_t)(oc0 + ct * 16) * HIDN, HIDN, k0);
            acc[0][ct] = wmma_bf16(a0, b, acc[0][ct]);
            acc[1][ct] = wmma_bf16(a1, b, acc[1][ct]);
        }
    }

    int l = threadIdx.x & 31, hi = l >> 4, cn = l & 15;
    #pragma unroll
    for (int rt = 0; rt < 2; ++rt)
        #pragma unroll
        for (int ct = 0; ct < 4; ++ct)
            #pragma unroll
            for (int v = 0; v < 8; ++v) {
                int gr = gr0 + rt * 16 + v + hi * 8;     // global row
                int o  = oc0 + ct * 16 + cn;             // output feature
                float val = (acc[rt][ct][v] + bias[o]) * scale;
                int b = gr >> 11, n = gr & (SEQ - 1);
                int head = o >> 7, d = o & (HDIM - 1);
                if (sel == 2)
                    out[(((size_t)b * HEADS + head) * HDIM + d) * SEQ + n] = f2bf(val);
                else
                    out[(((size_t)b * HEADS + head) * SEQ + n) * HDIM + d] = f2bf(val);
            }
}

// ---------------- kernel 3: flash attention ----------------
// grid = (B*HEADS, N/128). 8 waves/block share K and V^T tiles of 32 keys,
// staged into double-buffered LDS with async global->LDS loads (ASYNCcnt),
// overlapping the next tile's staging with the current tile's WMMA work.

__global__ __launch_bounds__(256) void attn_kernel(
    const bf16* __restrict__ Q, const bf16* __restrict__ K, const bf16* __restrict__ Vt,
    const int* __restrict__ mask, bf16* __restrict__ attended)
{
    __shared__ bf16 sK[2][32 * 128];    // keys x d      (double buffered)
    __shared__ bf16 sVt[2][128 * 32];   // d x keys      (double buffered)
    __shared__ bf16 sP[8 * 16 * 40];    // per-wave P staging (16 x 32, pitch 40)

    int bh = blockIdx.x;                // b*HEADS + h
    int b  = bh >> 3;
    int h  = bh & 7;
    int q0 = blockIdx.y * 128;
    int w  = threadIdx.x >> 5;
    int qw = q0 + w * 16;

    const bf16* Qb  = Q  + ((size_t)bh * SEQ + qw) * HDIM;
    const bf16* Kb  = K  + (size_t)bh * SEQ * HDIM;
    const bf16* Vtb = Vt + (size_t)bh * HDIM * SEQ;

    int l = threadIdx.x & 31, hi = l >> 4, cn = l & 15;

    // issue one stage: 512 x16B for K tile + 512 x16B for Vt tile = 4 asyncs/thread
    auto stage = [&](int j, int buf) {
        #pragma unroll
        for (int t0 = 0; t0 < 4; ++t0) {
            int t = threadIdx.x + t0 * 256;
            if (t < 512) {
                int key = t >> 4, c = t & 15;
                async_load16(Kb + (size_t)(j * 32 + key) * HDIM + c * 8,
                             &sK[buf][key * 128 + c * 8]);
            } else {
                int u = t - 512;
                int d = u >> 2, c = u & 3;
                async_load16(Vtb + (size_t)d * SEQ + j * 32 + c * 8,
                             &sVt[buf][d * 32 + c * 8]);
            }
        }
    };

    v16bf qf[4];
    #pragma unroll
    for (int kk = 0; kk < 4; ++kk) qf[kk] = load_a_frag(Qb, HDIM, kk * 32);

    v8f O[8];
    #pragma unroll
    for (int t = 0; t < 8; ++t) O[t] = zero8();
    float m[8], lsum[8];
    #pragma unroll
    for (int v = 0; v < 8; ++v) { m[v] = -1.0e30f; lsum[v] = 0.0f; }

    bf16* sPw = sP + w * 16 * 40;
    const int NJ = SEQ / 32;

    stage(0, 0);                               // prologue: fill buffer 0

    for (int j = 0; j < NJ; ++j) {
        int cur = j & 1;
        if (j + 1 < NJ) {
            stage(j + 1, cur ^ 1);             // prefetch next tile (other buffer)
            wait_async<4>();                   // in-order: current tile resident
        } else {
            wait_async<0>();
        }
        __syncthreads();                       // all waves' tiles visible

        const bf16* sKc  = sK[cur];
        const bf16* sVtc = sVt[cur];

        // S = Q * K^T for 32 keys (two 16-col C tiles)
        v8f S0 = zero8(), S1 = zero8();
        #pragma unroll
        for (int kk = 0; kk < 4; ++kk) {
            v16bf b0 = load_b_frag(sKc,            128, kk * 32);  // keys 0-15
            v16bf b1 = load_b_frag(sKc + 16 * 128, 128, kk * 32);  // keys 16-31
            S0 = wmma_bf16(qf[kk], b0, S0);
            S1 = wmma_bf16(qf[kk], b1, S1);
        }

        int mk0 = mask[b * SEQ + j * 32 + cn];
        int mk1 = mask[b * SEQ + j * 32 + 16 + cn];

        #pragma unroll
        for (int v = 0; v < 8; ++v) {
            float s0 = mk0 ? S0[v] : -1.0e9f;
            float s1 = mk1 ? S1[v] : -1.0e9f;
            float r = fmaxf(s0, s1);                 // row max over 32 keys
            r = fmaxf(r, __shfl_xor(r, 1));
            r = fmaxf(r, __shfl_xor(r, 2));
            r = fmaxf(r, __shfl_xor(r, 4));
            r = fmaxf(r, __shfl_xor(r, 8));
            float mn = fmaxf(m[v], r);
            float alpha = __expf(m[v] - mn);
            m[v] = mn;
            float p0 = __expf(s0 - mn);
            float p1 = __expf(s1 - mn);
            float rs = p0 + p1;                      // row sum
            rs += __shfl_xor(rs, 1); rs += __shfl_xor(rs, 2);
            rs += __shfl_xor(rs, 4); rs += __shfl_xor(rs, 8);
            lsum[v] = lsum[v] * alpha + rs;
            #pragma unroll
            for (int t = 0; t < 8; ++t) O[t][v] *= alpha;
            int row = v + hi * 8;
            sPw[row * 40 + cn]      = f2bf(p0);      // C layout -> LDS
            sPw[row * 40 + 16 + cn] = f2bf(p1);
        }

        // per-wave LDS ordering: P writes above, A-fragment reads below
        asm volatile("s_wait_dscnt 0" ::: "memory");

        v16bf pf = load_a_frag(sPw, 40, 0);          // P as 16x32 A fragment
        #pragma unroll
        for (int t = 0; t < 8; ++t) {
            v16bf bv = load_b_frag(sVtc + (size_t)(t * 16) * 32, 32, 0);
            O[t] = wmma_bf16(pf, bv, O[t]);          // O += P * V
        }
        __syncthreads();                             // done reading cur buffer
    }

    // normalize and write attended back to [B, N, HID] bf16
    #pragma unroll
    for (int v = 0; v < 8; ++v) {
        float inv = 1.0f / lsum[v];
        int row = qw + v + hi * 8;
        #pragma unroll
        for (int t = 0; t < 8; ++t) {
            int d = t * 16 + cn;
            attended[((size_t)b * SEQ + row) * HIDN + h * HDIM + d] = f2bf(O[t][v] * inv);
        }
    }
}

// ---------------- kernel 4: output projection + masked pooling ----------------

__global__ __launch_bounds__(256) void projpool_kernel(
    const bf16* __restrict__ A, const bf16* __restrict__ Wob,
    const int* __restrict__ mask, float* __restrict__ accPool)
{
    int w    = threadIdx.x >> 5;
    int tile = blockIdx.x * 8 + w;
    int gr0  = (tile >> 4) * 32;
    int oc0  = (tile & 15) * 64;

    v8f acc[2][4];
    #pragma unroll
    for (int r = 0; r < 2; ++r)
        #pragma unroll
        for (int c = 0; c < 4; ++c) acc[r][c] = zero8();

    for (int k0 = 0; k0 < HIDN; k0 += 32) {
        v16bf a0 = load_a_frag(A + (size_t)gr0 * HIDN,        HIDN, k0);
        v16bf a1 = load_a_frag(A + (size_t)(gr0 + 16) * HIDN, HIDN, k0);
        #pragma unroll
        for (int ct = 0; ct < 4; ++ct) {
            v16bf bfr = load_b_frag(Wob + (size_t)(oc0 + ct * 16) * HIDN, HIDN, k0);
            acc[0][ct] = wmma_bf16(a0, bfr, acc[0][ct]);
            acc[1][ct] = wmma_bf16(a1, bfr, acc[1][ct]);
        }
    }

    int l = threadIdx.x & 31, hi = l >> 4, cn = l & 15;
    int bI = gr0 >> 11;   // all 32 rows are in the same batch (2048 % 32 == 0)

    float mrow[2][8];
    #pragma unroll
    for (int rt = 0; rt < 2; ++rt)
        #pragma unroll
        for (int v = 0; v < 8; ++v) {
            int gr = gr0 + rt * 16 + v + hi * 8;
            mrow[rt][v] = (float)mask[bI * SEQ + (gr & (SEQ - 1))];
        }

    #pragma unroll
    for (int ct = 0; ct < 4; ++ct) {
        float colsum = 0.0f;
        #pragma unroll
        for (int rt = 0; rt < 2; ++rt)
            #pragma unroll
            for (int v = 0; v < 8; ++v) colsum += acc[rt][ct][v] * mrow[rt][v];
        colsum += __shfl_xor(colsum, 16);            // combine both half-rows
        if (l < 16) atomicAdd(&accPool[bI * HIDN + oc0 + ct * 16 + cn], colsum);
    }
}

// ---------------- kernel 5: finalize (divide by mask sum, add bias) ----------------

__global__ __launch_bounds__(256) void finalize_kernel(
    const float* __restrict__ accPool, const int* __restrict__ mask,
    const float* __restrict__ bo, float* __restrict__ out)
{
    __shared__ float red[256];
    int b = blockIdx.x;
    float s = 0.0f;
    for (int t = threadIdx.x; t < SEQ; t += 256) s += (float)mask[b * SEQ + t];
    red[threadIdx.x] = s;
    __syncthreads();
    for (int off = 128; off > 0; off >>= 1) {
        if (threadIdx.x < (unsigned)off) red[threadIdx.x] += red[threadIdx.x + off];
        __syncthreads();
    }
    float inv = 1.0f / red[0];
    for (int o = threadIdx.x; o < HIDN; o += 256)
        out[b * HIDN + o] = accPool[b * HIDN + o] * inv + bo[o];
}

// ---------------- launch ----------------

extern "C" void kernel_launch(void* const* d_in, const int* in_sizes, int n_in,
                              void* d_out, int out_size, void* d_ws, size_t ws_size,
                              hipStream_t stream)
{
    (void)in_sizes; (void)n_in; (void)out_size; (void)ws_size;
    const float* X    = (const float*)d_in[0];
    const int*   mask = (const int*)  d_in[1];
    const float* Wq   = (const float*)d_in[2];
    const float* bq   = (const float*)d_in[3];
    const float* Wk   = (const float*)d_in[4];
    const float* bk   = (const float*)d_in[5];
    const float* Wv   = (const float*)d_in[6];
    const float* bv   = (const float*)d_in[7];
    const float* Wo   = (const float*)d_in[8];
    const float* bo   = (const float*)d_in[9];
    float* out = (float*)d_out;

    char* ws = (char*)d_ws;
    size_t off = 0;
    auto alloc = [&](size_t bytes) -> void* {
        void* p = ws + off;
        off += (bytes + 255) & ~(size_t)255;
        return p;
    };

    bf16* Xb  = (bf16*)alloc((size_t)MROWS * HIDN * 2);
    bf16* Wqb = (bf16*)alloc((size_t)HIDN * HIDN * 2);
    bf16* Wkb = (bf16*)alloc((size_t)HIDN * HIDN * 2);
    bf16* Wvb = (bf16*)alloc((size_t)HIDN * HIDN * 2);
    bf16* Wob = (bf16*)alloc((size_t)HIDN * HIDN * 2);
    bf16* Qs  = (bf16*)alloc((size_t)MROWS * HIDN * 2);   // [B,H,N,D]
    bf16* Ks  = (bf16*)alloc((size_t)MROWS * HIDN * 2);   // [B,H,N,D]
    bf16* Vts = (bf16*)alloc((size_t)MROWS * HIDN * 2);   // [B,H,D,N] (transposed)
    bf16* Att = (bf16*)alloc((size_t)MROWS * HIDN * 2);   // [B,N,HID]
    float* accPool = (float*)alloc((size_t)BATCH * HIDN * 4);

    convert_kernel<<<1024, 256, 0, stream>>>(X, Wq, Wk, Wv, Wo,
                                             Xb, Wqb, Wkb, Wvb, Wob, accPool);

    qkv_kernel<<<dim3(512, 1, 3), 256, 0, stream>>>(Xb, Wqb, Wkb, Wvb,
                                                    bq, bk, bv, Qs, Ks, Vts);

    attn_kernel<<<dim3(BATCH * HEADS, SEQ / 128), 256, 0, stream>>>(Qs, Ks, Vts, mask, Att);

    projpool_kernel<<<512, 256, 0, stream>>>(Att, Wob, mask, accPool);

    finalize_kernel<<<BATCH, 256, 0, stream>>>(accPool, mask, bo, out);
}
